// AttentiveFPNet_90134183674516
// MI455X (gfx1250) — matrix operand order repaired and checked by reading, not compiled, not run on hardware
//
#include <hip/hip_runtime.h>

typedef unsigned short u16;
typedef __attribute__((ext_vector_type(16))) __bf16 v16bf;
typedef __attribute__((ext_vector_type(8)))  __bf16 v8bf;
typedef __attribute__((ext_vector_type(8)))  float   v8f;

constexpr int HDIM = 128;

#define SHUF16(lo, hi) __builtin_shufflevector(lo, hi, 0,1,2,3,4,5,6,7,8,9,10,11,12,13,14,15)

// epilogue modes
#define EPI_F32   0   // store f32
#define EPI_FB    1   // store f32 + bf16
#define EPI_BF    2   // store bf16
#define EPI_SCAT  3   // atomic scatter-add with per-row scale

// ---------------- device helpers ----------------
__device__ __forceinline__ u16 f2bf(float f) {
  unsigned u = __float_as_uint(f);
  u += 0x7FFFu + ((u >> 16) & 1u);   // round-to-nearest-even
  return (u16)(u >> 16);
}
__device__ __forceinline__ float bf2f(u16 b) {
  return __uint_as_float(((unsigned)b) << 16);
}
__device__ __forceinline__ float lrelu_f(float x) { return x > 0.f ? x : 0.01f * x; }
__device__ __forceinline__ float sigm_f(float x) { return 1.f / (1.f + __expf(-x)); }
__device__ __forceinline__ float wave_sum(float s) {
#pragma unroll
  for (int o = 16; o > 0; o >>= 1) s += __shfl_xor(s, o, 32);
  return s;
}
// order-preserving float<->uint map for atomic float max
__device__ __forceinline__ unsigned f2ord(float f) {
  unsigned u = __float_as_uint(f);
  return (u & 0x80000000u) ? ~u : (u ^ 0x80000000u);
}
__device__ __forceinline__ float ord2f(unsigned u) {
  unsigned v = (u & 0x80000000u) ? (u ^ 0x80000000u) : ~u;
  return __uint_as_float(v);
}

// ---------------- generic WMMA GEMM ----------------
// out = act( [A1(gather idx1) | A2(gather idx2)] @ Wt^T + bias )
// Wt pre-transposed bf16 [Nc][K1+K2]. A fragments are loaded ONCE into
// registers and reused for every 16-wide N tile (gathers are the bottleneck).
// EPI selects the (compile-time) epilogue; ACT=1 applies leaky-relu.
template <int NF1, int NF2, int EPI, int ACT>
__global__ __launch_bounds__(128) void gemm_wmma_kernel(
    const u16* __restrict__ A1, const int* __restrict__ idx1,
    const u16* __restrict__ A2, const int* __restrict__ idx2,
    const u16* __restrict__ Wt, const float* __restrict__ bias,
    float* __restrict__ outf, u16* __restrict__ outb,
    float* __restrict__ scat_out, const int* __restrict__ scat_idx,
    const float* __restrict__ scat_scale,
    int M, int Nc)
{
  constexpr int K1   = NF1 * 32;
  constexpr int K2   = NF2 * 32;
  constexpr int NF   = NF1 + NF2;
  constexpr int Ktot = K1 + K2;

  const int lane  = threadIdx.x & 31;
  const int wid   = threadIdx.x >> 5;          // wave32
  const int m0    = (blockIdx.x * 4 + wid) * 16;
  if (m0 >= M) return;                         // wave-uniform: EXEC stays all-ones
  const int l15   = lane & 15;
  const int khalf = lane >> 4;
  const bool full = (m0 + 16 <= M);

  int ar = m0 + l15; if (ar >= M) ar = M - 1;  // clamp tail rows for loads

  // A frag layout: lane = row M, element j -> K = f*32 + khalf*8 + j + (j>=8 ? 8 : 0)
  v16bf afrag[NF];
  {
    const int r1 = idx1 ? idx1[ar] : ar;
    const u16* a1 = A1 + (size_t)r1 * K1 + khalf * 8;
#pragma unroll
    for (int f = 0; f < NF1; ++f) {
      v8bf lo = *(const v8bf*)(a1 + f * 32);
      v8bf hi = *(const v8bf*)(a1 + f * 32 + 16);
      afrag[f] = SHUF16(lo, hi);
    }
  }
  if (NF2 > 0) {
    const int r2 = idx2 ? idx2[ar] : ar;
    const u16* a2 = A2 + (size_t)r2 * K2 + khalf * 8;
#pragma unroll
    for (int f = 0; f < NF2; ++f) {
      v8bf lo = *(const v8bf*)(a2 + f * 32);
      v8bf hi = *(const v8bf*)(a2 + f * 32 + 16);
      afrag[NF1 + f] = SHUF16(lo, hi);
    }
  }

  // scatter metadata is N-loop invariant: preload per-row idx/scale once
  int   sidx8[8];
  float sscl8[8];
  if (EPI == EPI_SCAT) {
#pragma unroll
    for (int j = 0; j < 8; ++j) {
      int m = m0 + khalf * 8 + j;
      int mc = m < M ? m : M - 1;
      sidx8[j] = scat_idx[mc];
      sscl8[j] = scat_scale[mc];
    }
  }

  // B frag layout: lane -> N = nt+(lane&15), K = f*32 + khalf*8 + j + (j>=8 ? 8 : 0)
  const u16* wbase = Wt + (size_t)l15 * Ktot + khalf * 8;
  for (int nt = 0; nt < Nc; nt += 16) {
    v8f acc = {0.f, 0.f, 0.f, 0.f, 0.f, 0.f, 0.f, 0.f};
    const u16* wb = wbase + (size_t)nt * Ktot;
#pragma unroll
    for (int f = 0; f < NF; ++f) {
      v8bf blo = *(const v8bf*)(wb + f * 32);
      v8bf bhi = *(const v8bf*)(wb + f * 32 + 16);
      v16bf b = SHUF16(blo, bhi);
      acc = __builtin_amdgcn_wmma_f32_16x16x32_bf16(false, afrag[f], false, b,
                                                    (short)0, acc, false, false);
    }
    const int col = nt + l15;
    const float bv = bias[col];
    if (full) {
#pragma unroll
      for (int j = 0; j < 8; ++j) {
        const int m = m0 + khalf * 8 + j;      // C layout: M = 8*lanehalf + vgpr j
        float v = acc[j] + bv;
        if (ACT) v = lrelu_f(v);
        if (EPI == EPI_F32) outf[(size_t)m * Nc + col] = v;
        if (EPI == EPI_FB)  { outf[(size_t)m * Nc + col] = v;
                              outb[(size_t)m * Nc + col] = f2bf(v); }
        if (EPI == EPI_BF)  outb[(size_t)m * Nc + col] = f2bf(v);
        if (EPI == EPI_SCAT)
          atomicAdd(&scat_out[(size_t)sidx8[j] * Nc + col], sscl8[j] * v);
      }
    } else {
#pragma unroll
      for (int j = 0; j < 8; ++j) {
        const int m = m0 + khalf * 8 + j;
        if (m < M) {
          float v = acc[j] + bv;
          if (ACT) v = lrelu_f(v);
          if (EPI == EPI_F32) outf[(size_t)m * Nc + col] = v;
          if (EPI == EPI_FB)  { outf[(size_t)m * Nc + col] = v;
                                outb[(size_t)m * Nc + col] = f2bf(v); }
          if (EPI == EPI_BF)  outb[(size_t)m * Nc + col] = f2bf(v);
          if (EPI == EPI_SCAT)
            atomicAdd(&scat_out[(size_t)sidx8[j] * Nc + col], sscl8[j] * v);
        }
      }
    }
  }
}

template <int NF1, int NF2, int EPI, int ACT>
static void launch_gemm(const u16* A1, const int* i1, const u16* A2, const int* i2,
                        const u16* Wt, const float* bias, float* of, u16* ob,
                        float* so, const int* si, const float* ss,
                        int M, int Nc, hipStream_t stream) {
  dim3 g((unsigned)((M + 63) / 64));
  gemm_wmma_kernel<NF1, NF2, EPI, ACT><<<g, 128, 0, stream>>>(
      A1, i1, A2, i2, Wt, bias, of, ob, so, si, ss, M, Nc);
}

// ---------------- small kernels ----------------
__global__ void fill_f32_kernel(float* p, float v, size_t n) {
  size_t i = (size_t)blockIdx.x * blockDim.x + threadIdx.x;
  if (i < n) p[i] = v;
}
__global__ void fill_u32_kernel(unsigned* p, unsigned v, size_t n) {
  size_t i = (size_t)blockIdx.x * blockDim.x + threadIdx.x;
  if (i < n) p[i] = v;
}
// f32 [M][Ks] -> bf16 [M][Kp] (zero-padded K)
__global__ void cvt_pad_kernel(const float* __restrict__ s, u16* __restrict__ d,
                               int M, int Ks, int Kp) {
  size_t i = (size_t)blockIdx.x * blockDim.x + threadIdx.x;
  if (i < (size_t)M * Kp) {
    int r = (int)(i / Kp), c = (int)(i % Kp);
    d[i] = (c < Ks) ? f2bf(s[(size_t)r * Ks + c]) : (u16)0;
  }
}
// W f32 [K][Nc] row-major -> Wt bf16 [Nc][Kp] (transposed, zero-padded K)
__global__ void prep_weight_kernel(const float* __restrict__ W, u16* __restrict__ Wt,
                                   int K, int Kp, int Nc) {
  size_t i = (size_t)blockIdx.x * blockDim.x + threadIdx.x;
  if (i < (size_t)Nc * Kp) {
    int c = (int)(i / Kp), k = (int)(i % Kp);
    Wt[i] = (k < K) ? f2bf(W[(size_t)k * Nc + c]) : (u16)0;
  }
}
// one wave per edge: score[e] = lrelu( msg[e]·Wal[0:H] + h[dst[e]]·Wal[H:2H] + b )
__global__ __launch_bounds__(256) void atom_score_kernel(
    const u16* __restrict__ msg, const float* __restrict__ h,
    const int* __restrict__ dst, const float* __restrict__ Wal,
    const float* __restrict__ bal, float* __restrict__ score, int E) {
  const int e = blockIdx.x * 8 + (threadIdx.x >> 5);
  const int lane = threadIdx.x & 31;
  if (e < E) {
    const u16* mp = msg + (size_t)e * HDIM;
    const float* hp = h + (size_t)dst[e] * HDIM;
    float s = 0.f;
#pragma unroll
    for (int k = lane; k < HDIM; k += 32)
      s += bf2f(mp[k]) * Wal[k] + hp[k] * Wal[HDIM + k];
    s = wave_sum(s);
    if (lane == 0) score[e] = lrelu_f(s + bal[0]);
  }
}
// one wave per node: sc[i] = lrelu( n[i]·Wal[0:H] + s[gid[i]]·Wal[H:2H] + b )
__global__ __launch_bounds__(256) void mol_score_kernel(
    const float* __restrict__ nf, const float* __restrict__ sf,
    const int* __restrict__ gid, const float* __restrict__ Wal,
    const float* __restrict__ bal, float* __restrict__ score, int N) {
  const int i = blockIdx.x * 8 + (threadIdx.x >> 5);
  const int lane = threadIdx.x & 31;
  if (i < N) {
    const float* np = nf + (size_t)i * HDIM;
    const float* sp = sf + (size_t)gid[i] * HDIM;
    float s = 0.f;
#pragma unroll
    for (int k = lane; k < HDIM; k += 32)
      s += np[k] * Wal[k] + sp[k] * Wal[HDIM + k];
    s = wave_sum(s);
    if (lane == 0) score[i] = lrelu_f(s + bal[0]);
  }
}
__global__ void seg_max_kernel(const float* __restrict__ score, const int* __restrict__ seg,
                               unsigned* __restrict__ smax, int n) {
  int i = blockIdx.x * blockDim.x + threadIdx.x;
  if (i < n) atomicMax(&smax[seg[i]], f2ord(score[i]));
}
__global__ void seg_exp_kernel(const float* __restrict__ score, const int* __restrict__ seg,
                               const unsigned* __restrict__ smax, float* __restrict__ att,
                               float* __restrict__ ssum, int n) {
  int i = blockIdx.x * blockDim.x + threadIdx.x;
  if (i < n) {
    float m = ord2f(smax[seg[i]]);
    float v = __expf(score[i] - m);
    att[i] = v;
    atomicAdd(&ssum[seg[i]], v);
  }
}
__global__ void seg_norm_kernel(float* __restrict__ att, const int* __restrict__ seg,
                                const float* __restrict__ ssum, int n) {
  int i = blockIdx.x * blockDim.x + threadIdx.x;
  if (i < n) att[i] /= ssum[seg[i]];
}
__global__ void elu_dual_kernel(float* __restrict__ x, u16* __restrict__ xb, size_t n) {
  size_t i = (size_t)blockIdx.x * blockDim.x + threadIdx.x;
  if (i < n) {
    float v = x[i];
    v = v > 0.f ? v : (__expf(v) - 1.f);
    x[i] = v; xb[i] = f2bf(v);
  }
}
__global__ void lrelu_dual_kernel(const float* __restrict__ x, float* __restrict__ y,
                                  u16* __restrict__ yb, size_t n) {
  size_t i = (size_t)blockIdx.x * blockDim.x + threadIdx.x;
  if (i < n) {
    float v = lrelu_f(x[i]);
    y[i] = v; yb[i] = f2bf(v);
  }
}
// GRU gates (torch order r,z,n) + outer ReLU
__global__ void gru_gate_kernel(const float* __restrict__ gi, const float* __restrict__ gh,
                                const float* __restrict__ hprev, float* __restrict__ out,
                                u16* __restrict__ outb, int M) {
  size_t i = (size_t)blockIdx.x * blockDim.x + threadIdx.x;
  if (i < (size_t)M * HDIM) {
    int r = (int)(i / HDIM), c = (int)(i % HDIM);
    const float* gir = gi + (size_t)r * 3 * HDIM;
    const float* ghr = gh + (size_t)r * 3 * HDIM;
    float rg = sigm_f(gir[c] + ghr[c]);
    float zg = sigm_f(gir[HDIM + c] + ghr[HDIM + c]);
    float ng = tanhf(gir[2 * HDIM + c] + rg * ghr[2 * HDIM + c]);
    float v = (1.f - zg) * ng + zg * hprev[i];
    v = v > 0.f ? v : 0.f;
    out[i] = v;
    if (outb) outb[i] = f2bf(v);
  }
}
__global__ void scatter_sum_kernel(const float* __restrict__ x, const int* __restrict__ gid,
                                   float* __restrict__ out, int N) {
  size_t i = (size_t)blockIdx.x * blockDim.x + threadIdx.x;
  if (i < (size_t)N * HDIM) {
    int r = (int)(i / HDIM), c = (int)(i % HDIM);
    atomicAdd(&out[(size_t)gid[r] * HDIM + c], x[i]);
  }
}
// one wave per graph
__global__ __launch_bounds__(256) void final_kernel(
    const float* __restrict__ sn, const float* __restrict__ Wp,
    const float* __restrict__ bp, float* __restrict__ out, int G) {
  const int g = blockIdx.x * 8 + (threadIdx.x >> 5);
  const int lane = threadIdx.x & 31;
  if (g < G) {
    const float* p = sn + (size_t)g * HDIM;
    float s = 0.f;
#pragma unroll
    for (int k = lane; k < HDIM; k += 32) s += p[k] * Wp[k];
    s = wave_sum(s);
    if (lane == 0) out[g] = s + bp[0];
  }
}

// ---------------- host orchestration ----------------
static inline dim3 grid1(size_t n, int t) { return dim3((unsigned)((n + (size_t)t - 1) / (size_t)t)); }

extern "C" void kernel_launch(void* const* d_in, const int* in_sizes, int n_in,
                              void* d_out, int out_size, void* d_ws, size_t ws_size,
                              hipStream_t stream) {
  const float* node = (const float*)d_in[0];
  const float* edge = (const float*)d_in[1];
  const int*   src  = (const int*)d_in[2];
  const int*   dst  = (const int*)d_in[3];
  const int*   gid  = (const int*)d_in[4];
  const float* Wn0 = (const float*)d_in[5];  const float* bn0 = (const float*)d_in[6];
  const float* We0 = (const float*)d_in[7];  const float* be0 = (const float*)d_in[8];
  const float* aWn = (const float*)d_in[9];  const float* abn = (const float*)d_in[10];
  const float* aWe = (const float*)d_in[11]; const float* abe = (const float*)d_in[12];
  const float* aWal= (const float*)d_in[13]; const float* abal= (const float*)d_in[14];
  const float* aWat= (const float*)d_in[15]; const float* abat= (const float*)d_in[16];
  const float* aWih= (const float*)d_in[17]; const float* abih= (const float*)d_in[18];
  const float* aWhh= (const float*)d_in[19]; const float* abhh= (const float*)d_in[20];
  const float* mWal= (const float*)d_in[21]; const float* mbal= (const float*)d_in[22];
  const float* mWat= (const float*)d_in[23]; const float* mbat= (const float*)d_in[24];
  const float* mWih= (const float*)d_in[25]; const float* mbih= (const float*)d_in[26];
  const float* mWhh= (const float*)d_in[27]; const float* mbhh= (const float*)d_in[28];
  const float* Wp  = (const float*)d_in[29]; const float* bp  = (const float*)d_in[30];
  float* out = (float*)d_out;

  const int FNk = 64, FEk = 16, FEp = 32;
  const int N = in_sizes[0] / FNk;
  const int E = in_sizes[1] / FEk;
  const int G = out_size;
  const int D = in_sizes[9]  / (HDIM * HDIM);
  const int L = in_sizes[23] / (HDIM * HDIM);

  char* ws = (char*)d_ws;
  size_t off = 0;
  auto alloc = [&](size_t bytes) -> char* {
    char* p = ws + off;
    off += (bytes + 255) & ~(size_t)255;
    return p;
  };

  // transposed bf16 weights
  u16* Wn0t  = (u16*)alloc((size_t)HDIM * FNk * 2);
  u16* We0t  = (u16*)alloc((size_t)HDIM * FEp * 2);
  u16* aWnT  = (u16*)alloc((size_t)D * HDIM * HDIM * 2);
  u16* aWeT  = (u16*)alloc((size_t)D * HDIM * 2 * HDIM * 2);
  u16* aWatT = (u16*)alloc((size_t)D * HDIM * HDIM * 2);
  u16* aWihT = (u16*)alloc((size_t)D * 3 * HDIM * HDIM * 2);
  u16* aWhhT = (u16*)alloc((size_t)D * 3 * HDIM * HDIM * 2);
  u16* mWatT = (u16*)alloc((size_t)L * HDIM * HDIM * 2);
  u16* mWihT = (u16*)alloc((size_t)L * 3 * HDIM * HDIM * 2);
  u16* mWhhT = (u16*)alloc((size_t)L * 3 * HDIM * HDIM * 2);
  // activations / scratch
  u16*   node_bf = (u16*)alloc((size_t)N * FNk * 2);
  u16*   edge_bf = (u16*)alloc((size_t)E * FEp * 2);
  float* n_f     = (float*)alloc((size_t)N * HDIM * 4);
  u16*   n_bf    = (u16*)alloc((size_t)N * HDIM * 2);
  u16*   e_bf    = (u16*)alloc((size_t)E * HDIM * 2);
  float* h_f     = (float*)alloc((size_t)N * HDIM * 4);
  u16*   h_bf    = (u16*)alloc((size_t)N * HDIM * 2);
  u16*   msg_bf  = (u16*)alloc((size_t)E * HDIM * 2);
  float* ctx     = (float*)alloc((size_t)N * HDIM * 4);
  u16*   ctx_bf  = (u16*)alloc((size_t)N * HDIM * 2);
  float* gi      = (float*)alloc((size_t)N * 3 * HDIM * 4);
  float* gh      = (float*)alloc((size_t)N * 3 * HDIM * 4);
  float* score   = (float*)alloc((size_t)E * 4);
  float* att     = (float*)alloc((size_t)E * 4);
  unsigned* smax = (unsigned*)alloc((size_t)N * 4);
  float* ssum    = (float*)alloc((size_t)N * 4);
  float* sn      = (float*)alloc((size_t)G * HDIM * 4);
  float* s_f     = (float*)alloc((size_t)G * HDIM * 4);
  u16*   s_bf    = (u16*)alloc((size_t)G * HDIM * 2);
  float* sctx    = (float*)alloc((size_t)G * HDIM * 4);
  u16*   sctx_bf = (u16*)alloc((size_t)G * HDIM * 2);

  auto prep = [&](const float* W, u16* Wt, int K, int Kp, int Nc) {
    size_t n = (size_t)Nc * Kp;
    prep_weight_kernel<<<grid1(n, 256), 256, 0, stream>>>(W, Wt, K, Kp, Nc);
  };
  auto fillf = [&](float* p, float v, size_t n) {
    fill_f32_kernel<<<grid1(n, 256), 256, 0, stream>>>(p, v, n);
  };
  auto fillu = [&](unsigned* p, unsigned v, size_t n) {
    fill_u32_kernel<<<grid1(n, 256), 256, 0, stream>>>(p, v, n);
  };

  // ---- weight prep (transpose + bf16) ----
  prep(Wn0, Wn0t, FNk, FNk, HDIM);
  prep(We0, We0t, FEk, FEp, HDIM);
  for (int d = 0; d < D; ++d) {
    prep(aWn  + (size_t)d * HDIM * HDIM,     aWnT  + (size_t)d * HDIM * HDIM,     HDIM,     HDIM,     HDIM);
    prep(aWe  + (size_t)d * 2 * HDIM * HDIM, aWeT  + (size_t)d * 2 * HDIM * HDIM, 2 * HDIM, 2 * HDIM, HDIM);
    prep(aWat + (size_t)d * HDIM * HDIM,     aWatT + (size_t)d * HDIM * HDIM,     HDIM,     HDIM,     HDIM);
    prep(aWih + (size_t)d * HDIM * 3 * HDIM, aWihT + (size_t)d * 3 * HDIM * HDIM, HDIM,     HDIM,     3 * HDIM);
    prep(aWhh + (size_t)d * HDIM * 3 * HDIM, aWhhT + (size_t)d * 3 * HDIM * HDIM, HDIM,     HDIM,     3 * HDIM);
  }
  for (int l = 0; l < L; ++l) {
    prep(mWat + (size_t)l * HDIM * HDIM,     mWatT + (size_t)l * HDIM * HDIM,     HDIM, HDIM, HDIM);
    prep(mWih + (size_t)l * HDIM * 3 * HDIM, mWihT + (size_t)l * 3 * HDIM * HDIM, HDIM, HDIM, 3 * HDIM);
    prep(mWhh + (size_t)l * HDIM * 3 * HDIM, mWhhT + (size_t)l * 3 * HDIM * HDIM, HDIM, HDIM, 3 * HDIM);
  }

  // ---- input embedding ----
  cvt_pad_kernel<<<grid1((size_t)N * FNk, 256), 256, 0, stream>>>(node, node_bf, N, FNk, FNk);
  cvt_pad_kernel<<<grid1((size_t)E * FEp, 256), 256, 0, stream>>>(edge, edge_bf, E, FEk, FEp);
  launch_gemm<2, 0, EPI_FB, 1>(node_bf, nullptr, nullptr, nullptr, Wn0t, bn0,
                               n_f, n_bf, nullptr, nullptr, nullptr, N, HDIM, stream);
  launch_gemm<1, 0, EPI_BF, 1>(edge_bf, nullptr, nullptr, nullptr, We0t, be0,
                               nullptr, e_bf, nullptr, nullptr, nullptr, E, HDIM, stream);

  // ---- atom message passing ----
  for (int d = 0; d < D; ++d) {
    launch_gemm<4, 0, EPI_FB, 1>(n_bf, nullptr, nullptr, nullptr,
                                 aWnT + (size_t)d * HDIM * HDIM, abn + (size_t)d * HDIM,
                                 h_f, h_bf, nullptr, nullptr, nullptr, N, HDIM, stream);
    launch_gemm<4, 4, EPI_BF, 1>(h_bf, src, e_bf, nullptr,
                                 aWeT + (size_t)d * 2 * HDIM * HDIM, abe + (size_t)d * HDIM,
                                 nullptr, msg_bf, nullptr, nullptr, nullptr, E, HDIM, stream);
    atom_score_kernel<<<grid1((size_t)E, 8), 256, 0, stream>>>(
        msg_bf, h_f, dst, aWal + (size_t)d * 2 * HDIM, abal + d, score, E);
    fillu(smax, 0u, N);
    fillf(ssum, 0.f, N);
    seg_max_kernel<<<grid1(E, 256), 256, 0, stream>>>(score, dst, smax, E);
    seg_exp_kernel<<<grid1(E, 256), 256, 0, stream>>>(score, dst, smax, att, ssum, E);
    seg_norm_kernel<<<grid1(E, 256), 256, 0, stream>>>(att, dst, ssum, E);
    fillf(ctx, 0.f, (size_t)N * HDIM);
    launch_gemm<4, 0, EPI_SCAT, 0>(msg_bf, nullptr, nullptr, nullptr,
                                   aWatT + (size_t)d * HDIM * HDIM, abat + (size_t)d * HDIM,
                                   nullptr, nullptr, ctx, dst, att, E, HDIM, stream);
    elu_dual_kernel<<<grid1((size_t)N * HDIM, 256), 256, 0, stream>>>(ctx, ctx_bf, (size_t)N * HDIM);
    launch_gemm<4, 0, EPI_F32, 0>(ctx_bf, nullptr, nullptr, nullptr,
                                  aWihT + (size_t)d * 3 * HDIM * HDIM, abih + (size_t)d * 3 * HDIM,
                                  gi, nullptr, nullptr, nullptr, nullptr, N, 3 * HDIM, stream);
    launch_gemm<4, 0, EPI_F32, 0>(h_bf, nullptr, nullptr, nullptr,
                                  aWhhT + (size_t)d * 3 * HDIM * HDIM, abhh + (size_t)d * 3 * HDIM,
                                  gh, nullptr, nullptr, nullptr, nullptr, N, 3 * HDIM, stream);
    gru_gate_kernel<<<grid1((size_t)N * HDIM, 256), 256, 0, stream>>>(gi, gh, h_f, n_f, n_bf, N);
  }

  // ---- molecule readout ----
  fillf(sn, 0.f, (size_t)G * HDIM);
  scatter_sum_kernel<<<grid1((size_t)N * HDIM, 256), 256, 0, stream>>>(n_f, gid, sn, N);
  for (int l = 0; l < L; ++l) {
    lrelu_dual_kernel<<<grid1((size_t)G * HDIM, 256), 256, 0, stream>>>(sn, s_f, s_bf, (size_t)G * HDIM);
    mol_score_kernel<<<grid1((size_t)N, 8), 256, 0, stream>>>(
        n_f, s_f, gid, mWal + (size_t)l * 2 * HDIM, mbal + l, score, N);
    fillu(smax, 0u, G);
    fillf(ssum, 0.f, G);
    seg_max_kernel<<<grid1(N, 256), 256, 0, stream>>>(score, gid, smax, N);
    seg_exp_kernel<<<grid1(N, 256), 256, 0, stream>>>(score, gid, smax, att, ssum, N);
    seg_norm_kernel<<<grid1(N, 256), 256, 0, stream>>>(att, gid, ssum, N);
    fillf(sctx, 0.f, (size_t)G * HDIM);
    launch_gemm<4, 0, EPI_SCAT, 0>(n_bf, nullptr, nullptr, nullptr,
                                   mWatT + (size_t)l * HDIM * HDIM, mbat + (size_t)l * HDIM,
                                   nullptr, nullptr, sctx, gid, att, N, HDIM, stream);
    elu_dual_kernel<<<grid1((size_t)G * HDIM, 256), 256, 0, stream>>>(sctx, sctx_bf, (size_t)G * HDIM);
    launch_gemm<4, 0, EPI_F32, 0>(sctx_bf, nullptr, nullptr, nullptr,
                                  mWihT + (size_t)l * 3 * HDIM * HDIM, mbih + (size_t)l * 3 * HDIM,
                                  gi, nullptr, nullptr, nullptr, nullptr, G, 3 * HDIM, stream);
    launch_gemm<4, 0, EPI_F32, 0>(s_bf, nullptr, nullptr, nullptr,
                                  mWhhT + (size_t)l * 3 * HDIM * HDIM, mbhh + (size_t)l * 3 * HDIM,
                                  gh, nullptr, nullptr, nullptr, nullptr, G, 3 * HDIM, stream);
    gru_gate_kernel<<<grid1((size_t)G * HDIM, 256), 256, 0, stream>>>(gi, gh, s_f, sn, nullptr, G);
  }

  final_kernel<<<grid1((size_t)G, 8), 256, 0, stream>>>(sn, Wp, bp, out, G);
  (void)n_in; (void)ws_size;
}
